// cat3Head_49478023250675
// MI455X (gfx1250) — matrix operand order repaired and checked by reading, not compiled, not run on hardware
//
#include <hip/hip_runtime.h>

typedef __attribute__((ext_vector_type(16))) _Float16 v16h;
typedef __attribute__((ext_vector_type(8)))  float    v8f;

#define DIM      128
#define EIN      64
#define TILE     16
#define NWAVES   24          // one wave per 16-column tile of the 384 output cols
#define THREADS  (NWAVES*32)
#define TSTRIDE  136         // padded LDS row stride (floats): 16B-aligned, bank-skewed
#define ASTRIDE  28          // accT row stride: 112B rows, 16B-aligned, conflict-free

__global__ __launch_bounds__(THREADS)
void cat3head_edge_kernel(const float* __restrict__ x,
                          const long long* __restrict__ ei,
                          const float* __restrict__ attr3,
                          const float* __restrict__ Ww,
                          const float* __restrict__ Wb,
                          float* __restrict__ out,
                          int E, int numTiles)
{
    __shared__ __align__(16) float t0s[TILE * TSTRIDE];
    __shared__ __align__(16) float t1s[TILE * TSTRIDE];
    __shared__ __align__(16) float accT[TILE * ASTRIDE];  // [edge][wave]
    __shared__ float wbs[EIN];

    const int tid  = threadIdx.x;
    const int lane = tid & 31;
    const int w    = tid >> 5;           // wave id == 16-wide N-tile id (0..23)
    const int el   = lane & 15;          // A-row (W col) AND B-col (edge) for this lane
    const int hi   = (lane >= 16) ? 1 : 0;

    if (tid < EIN) wbs[tid] = Wb[tid];   // covered by first loop-top barrier

    // ---- A fragments: W^T tile, rows = this wave's 16 output columns, K = 64 ----
    // A layout: lane<16 row el, K {kb..kb+15} contiguous in elems 0-15;
    // kb = 0 (lane<16) / 16 (lane>=16); fragment 1 adds 32. Loaded ONCE, register-resident.
    const int ng  = w * TILE + el;       // global output column in [0,384)
    const int kbA = hi ? 16 : 0;
    v16h a0, a1;
    #pragma unroll
    for (int j = 0; j < 8; ++j) {
        a0[j]     = (_Float16)Ww[(kbA + j) * 384 + ng];
        a0[8 + j] = (_Float16)Ww[(kbA + 8 + j) * 384 + ng];
        a1[j]     = (_Float16)Ww[(32 + kbA + j) * 384 + ng];
        a1[8 + j] = (_Float16)Ww[(32 + kbA + 8 + j) * 384 + ng];
    }

    // Branchless yhat blend coefficients (wave-uniform selector)
    const int   sel = w >> 3;                       // 0: mean, 1: prod, 2: sq-diff
    const float cm  = (sel == 0) ? 0.5f : 0.f;
    const float cp  = (sel == 1) ? 1.f  : 0.f;
    const float cs  = (sel == 2) ? 1.f  : 0.f;
    const int   mbase = hi ? 8 : 0;                 // C rows for this lane half
    const int   d0    = ((w & 7) << 4) + mbase;     // contiguous dims d0..d0+7
    const int   kbB   = hi ? 16 : 0;                // B layout K base

    for (int t = blockIdx.x; t < numTiles; t += gridDim.x) {
        const int base = t * TILE;
        __syncthreads();  // previous tile's LDS readers done

        // ---- prefetch next tile's attr3 rows (256B row = 2 lines), one wave ----
        if (w == 1 && !hi) {
            const int ne = base + (int)gridDim.x * TILE + el;
            if (ne < E) {
                const float* nrow = attr3 + (size_t)ne * EIN;
                __builtin_prefetch(nrow, 0, 0);
                __builtin_prefetch(nrow + 32, 0, 0);
            }
        }

        // ---- cooperatively stage gathered node features t0,t1 into LDS ----
        for (int i = tid; i < TILE * (DIM / 4); i += THREADS) {
            const int m  = i >> 5;            // edge row in tile
            const int c4 = i & 31;            // float4 column
            int e = base + m; if (e >= E) e = E - 1;
            const int i0 = (int)ei[e];
            const int i1 = (int)ei[(size_t)E + e];
            const float4 v0 = ((const float4*)(x + (size_t)i0 * DIM))[c4];
            const float4 v1 = ((const float4*)(x + (size_t)i1 * DIM))[c4];
            *(float4*)&t0s[m * TSTRIDE + c4 * 4] = v0;
            *(float4*)&t1s[m * TSTRIDE + c4 * 4] = v1;
        }

        // ---- B fragments: attr3^T (64 x 16 edges), contiguous K runs, f32 -> f16 ----
        {
            int e = base + el; if (e >= E) e = E - 1;
            const float* arow = attr3 + (size_t)e * EIN;
            v16h b0, b1;
            float af[32];
            #pragma unroll
            for (int j = 0; j < 16; ++j) {
                af[j]      = arow[kbB + j];                // K = 0..31 half
                af[16 + j] = arow[32 + kbB + j];           // K = 32..63 half
                b0[j] = (_Float16)af[j];
                b1[j] = (_Float16)af[16 + j];
            }

            // bias = attr3[e] . Wb, folded into wave 0 from already-loaded registers;
            // lane covers its 32-K half, the xor-16 shuffle below merges the halves.
            float s = 0.f;
            if (w == 0) {
                #pragma unroll
                for (int j = 0; j < 16; ++j) {
                    s = fmaf(af[j],      wbs[kbB + j],      s);
                    s = fmaf(af[16 + j], wbs[32 + kbB + j], s);
                }
            }
            __syncthreads();  // t0s/t1s staged

            // P'[n, e] = W^T[n, :64] @ attr3^T[:64, e]  -> lane = edge, VGPR = n
            v8f c = {0.f, 0.f, 0.f, 0.f, 0.f, 0.f, 0.f, 0.f};
            c = __builtin_amdgcn_wmma_f32_16x16x32_f16(false, a0, false, b0,
                                                       (short)0, c, false, false);
            c = __builtin_amdgcn_wmma_f32_16x16x32_f16(false, a1, false, b1,
                                                       (short)0, c, false, false);

            // yhat dims for this lane are contiguous: two b128 LDS loads per operand
            float ta[8], tb[8];
            *(float4*)&ta[0] = *(const float4*)&t0s[el * TSTRIDE + d0];
            *(float4*)&ta[4] = *(const float4*)&t0s[el * TSTRIDE + d0 + 4];
            *(float4*)&tb[0] = *(const float4*)&t1s[el * TSTRIDE + d0];
            *(float4*)&tb[4] = *(const float4*)&t1s[el * TSTRIDE + d0 + 4];

            // fold the 8 output-column products inside the lane (single scalar)
            #pragma unroll
            for (int r = 0; r < 8; ++r) {
                const float dd = ta[r] - tb[r];
                const float y  = cm * (ta[r] + tb[r]) + cp * (ta[r] * tb[r]) + cs * (dd * dd);
                s = fmaf(c[r], y, s);
            }
            // merge the two lane halves (n and bias K halves): ONE shuffle
            s += __shfl_xor(s, 16, 32);
            if (!hi) accT[el * ASTRIDE + w] = s;
        }
        __syncthreads();

        // ---- cross-wave reduce (vectorized b128 reads), write 16 outputs ----
        if (tid < TILE) {
            const int e = base + tid;
            if (e < E) {
                const float4* row = (const float4*)&accT[tid * ASTRIDE];
                float s = 0.f;
                #pragma unroll
                for (int q = 0; q < NWAVES / 4; ++q) {
                    const float4 v = row[q];
                    s += (v.x + v.y) + (v.z + v.w);
                }
                out[e] = s;
            }
        }
    }
}

extern "C" void kernel_launch(void* const* d_in, const int* in_sizes, int n_in,
                              void* d_out, int out_size, void* d_ws, size_t ws_size,
                              hipStream_t stream)
{
    (void)n_in; (void)out_size; (void)d_ws; (void)ws_size;
    const float*     x     = (const float*)d_in[0];        // [N,128] f32
    const long long* ei    = (const long long*)d_in[1];    // [2,E] i64
    const float*     attr3 = (const float*)d_in[2];        // [E,64] f32
    const float*     Ww    = (const float*)d_in[5];        // [64,384] f32
    const float*     Wb    = (const float*)d_in[6];        // [64,1] f32
    float* out = (float*)d_out;                            // [E] f32

    const int E = in_sizes[2] / EIN;
    const int numTiles = (E + TILE - 1) / TILE;
    const int blocks = numTiles < 2048 ? numTiles : 2048;
    cat3head_edge_kernel<<<blocks, THREADS, 0, stream>>>(x, ei, attr3, Ww, Wb,
                                                         out, E, numTiles);
}
